// MultiHeadedAttention_33809982554179
// MI455X (gfx1250) — matrix operand order repaired and checked by reading, not compiled, not run on hardware
//
#include <hip/hip_runtime.h>
#include <hip/hip_bf16.h>

typedef __attribute__((ext_vector_type(16))) __bf16 v16bf;
typedef __attribute__((ext_vector_type(8)))  __bf16 v8bf;
typedef __attribute__((ext_vector_type(8)))  float  v8f;
typedef __attribute__((ext_vector_type(4)))  int    v4i_;

#define B_   2
#define S_   2048
#define D_   1024
#define H_   16
#define DK_  64
#define MTOT (B_ * S_)

// ---------------------------------------------------------------------------
// CDNA5 async copy global->LDS (ASYNCcnt-tracked, bypasses VGPRs).
// Builtin signature (from hipcc diagnostic): param0 = addrspace(1) v4i*,
// param1 = addrspace(3) v4i*, then imm offset, imm cpol.
// LDS operand: low 32 bits of a generic LDS address == LDS byte offset
// (aperture layout, ISA 10.2), so an integer round-trip forms the as(3) ptr.
// ---------------------------------------------------------------------------
#if __has_builtin(__builtin_amdgcn_global_load_async_to_lds_b128)
#define USE_ASYNC_LDS 1
#else
#define USE_ASYNC_LDS 0
#endif

__device__ __forceinline__ void async_copy16(void* lds, const void* gmem) {
#if USE_ASYNC_LDS
  __builtin_amdgcn_global_load_async_to_lds_b128(
      (__attribute__((address_space(1))) v4i_*)(size_t)gmem,
      (__attribute__((address_space(3))) v4i_*)(unsigned)(size_t)lds,
      0, 0);
#else
  *reinterpret_cast<v8bf*>(lds) = *reinterpret_cast<const v8bf*>(gmem);
#endif
}

__device__ __forceinline__ void async_wait0() {
#if USE_ASYNC_LDS
#if __has_builtin(__builtin_amdgcn_s_wait_asynccnt)
  __builtin_amdgcn_s_wait_asynccnt(0);
#else
  asm volatile("s_wait_asynccnt 0" ::: "memory");
#endif
#endif
}

// ---------------------------------------------------------------------------
// f32 -> bf16 convert (one-time; all GEMMs then run the WMMA bf16 path)
// ---------------------------------------------------------------------------
__global__ void cvt_f32_bf16(const float* __restrict__ src,
                             __bf16* __restrict__ dst, int n) {
  int i = blockIdx.x * blockDim.x + threadIdx.x;
  if (i < n) dst[i] = (__bf16)src[i];
}

// ---------------------------------------------------------------------------
// Tiled WMMA GEMM:  C[M,N] = A[M,K] * W[N,K]^T + bias[N]
// MODE 0: store bf16 into [B,H,S,DK] head layout (QKV projections)
// MODE 1: store f32 flat [M,N] (output projection -> d_out)
// 256 threads (8 waves); tile 128x128, K-step 32; DOUBLE-BUFFERED LDS with
// async global->LDS fills overlapping the WMMA work on the other buffer.
// ---------------------------------------------------------------------------
template<int MODE>
__global__ __launch_bounds__(256) void gemm_bf16_wmma(
    const __bf16* __restrict__ A, const __bf16* __restrict__ W,
    const float* __restrict__ bias, void* __restrict__ out,
    int M, int N, int K)
{
  constexpr int BM = 128, BN = 128, BK = 32, LDT = BK + 8;  // padded stride
  __shared__ alignas(16) __bf16 sA[2][BM * LDT];
  __shared__ alignas(16) __bf16 sW[2][BN * LDT];

  const int tid  = threadIdx.x;
  const int lane = tid & 31;
  const int wid  = tid >> 5;
  const int hi   = lane >> 4;     // 0: lanes 0-15, 1: lanes 16-31
  const int l15  = lane & 15;
  const int wm   = (wid >> 2) * 64;   // wave M offset in tile
  const int wn   = (wid & 3) * 32;    // wave N offset in tile
  const int m0   = blockIdx.y * BM;
  const int n0   = blockIdx.x * BN;

  const int ldrow = tid >> 1;         // 0..127
  const int ldcol = (tid & 1) * 16;   // 0 or 16 (bf16 elements)

  auto loadTiles = [&](int kt, int buf) {
    const __bf16* ga = A + (size_t)(m0 + ldrow) * K + kt + ldcol;
    const __bf16* gw = W + (size_t)(n0 + ldrow) * K + kt + ldcol;
    async_copy16(&sA[buf][ldrow * LDT + ldcol],     ga);
    async_copy16(&sA[buf][ldrow * LDT + ldcol + 8], ga + 8);
    async_copy16(&sW[buf][ldrow * LDT + ldcol],     gw);
    async_copy16(&sW[buf][ldrow * LDT + ldcol + 8], gw + 8);
  };

  v8f acc[4][2];
#pragma unroll
  for (int a = 0; a < 4; ++a)
#pragma unroll
    for (int c = 0; c < 2; ++c)
#pragma unroll
      for (int r = 0; r < 8; ++r) acc[a][c][r] = 0.f;

  // prologue: fill buffer 0
  loadTiles(0, 0);
  async_wait0();
  __syncthreads();

  const int nkt = K / BK;
  for (int t = 0; t < nkt; ++t) {
    const int cur = t & 1;
    // software pipeline: kick off next tile while this one computes
    if (t + 1 < nkt) loadTiles((t + 1) * BK, cur ^ 1);

    // ---- A fragments: 16x32 bf16, row = l15, K split per lane half ----
    v16bf af[4], bfr[2];
#pragma unroll
    for (int ms = 0; ms < 4; ++ms) {
      const __bf16* base = &sA[cur][(wm + ms * 16 + l15) * LDT];
      v8bf lo = *reinterpret_cast<const v8bf*>(base + hi * 8);
      v8bf hh = *reinterpret_cast<const v8bf*>(base + 16 + hi * 8);
#pragma unroll
      for (int j = 0; j < 8; ++j) { af[ms][j] = lo[j]; af[ms][8 + j] = hh[j]; }
    }
    // ---- B fragments: 32x16 bf16, col = l15, rows K = hi*8 + i and 16 + hi*8 + i ----
#pragma unroll
    for (int ns = 0; ns < 2; ++ns) {
      const __bf16* base = &sW[cur][(wn + ns * 16 + l15) * LDT + hi * 8];
      v8bf b0 = *reinterpret_cast<const v8bf*>(base);
      v8bf b1 = *reinterpret_cast<const v8bf*>(base + 16);
#pragma unroll
      for (int j = 0; j < 8; ++j) { bfr[ns][j] = b0[j]; bfr[ns][8 + j] = b1[j]; }
    }
#pragma unroll
    for (int ms = 0; ms < 4; ++ms)
#pragma unroll
      for (int ns = 0; ns < 2; ++ns)
        acc[ms][ns] = __builtin_amdgcn_wmma_f32_16x16x32_bf16(
            false, af[ms], false, bfr[ns], (short)0, acc[ms][ns], false, false);

    // ensure next tile landed (our own async ops) then rendezvous
    async_wait0();
    __syncthreads();
  }

  // ---- epilogue: bias + store (C layout: row = hi*8 + r, col = l15) ----
#pragma unroll
  for (int ms = 0; ms < 4; ++ms)
#pragma unroll
    for (int ns = 0; ns < 2; ++ns) {
      const int n = n0 + wn + ns * 16 + l15;
      const float bn = bias[n];
#pragma unroll
      for (int r = 0; r < 8; ++r) {
        const int m = m0 + wm + ms * 16 + hi * 8 + r;
        const float v = acc[ms][ns][r] + bn;
        if constexpr (MODE == 0) {
          const int bb = m >> 11, s = m & (S_ - 1);
          const int hd = n >> 6,  dk = n & (DK_ - 1);
          reinterpret_cast<__bf16*>(out)
              [(((size_t)bb * H_ + hd) * S_ + s) * DK_ + dk] = (__bf16)v;
        } else {
          reinterpret_cast<float*>(out)[(size_t)m * N + n] = v;
        }
      }
    }
}

// ---------------------------------------------------------------------------
// Flash attention: grid (S/64, B*H), 128 threads (4 waves, 16 q-rows each).
// K tile async-copied to LDS; V tile transposed into LDS; scores & P*V via
// WMMA bf16; online softmax with 16-lane shfl reductions in WMMA C layout.
// Next K/V tiles are prefetched (global_prefetch_b8) while computing.
// ---------------------------------------------------------------------------
__global__ __launch_bounds__(128) void flash_attn_wmma(
    const __bf16* __restrict__ Q, const __bf16* __restrict__ K,
    const __bf16* __restrict__ V, const unsigned char* __restrict__ mask,
    __bf16* __restrict__ ctx)
{
  constexpr int VSTR = DK_ + 8;  // 72, padded LDS stride
  __shared__ alignas(16) __bf16 sK [64 * VSTR];
  __shared__ alignas(16) __bf16 sVT[DK_ * VSTR];   // [dk][key]
  __shared__ alignas(16) __bf16 sP [4][16 * 64];   // per-wave P stage

  const int tid = threadIdx.x, lane = tid & 31, wid = tid >> 5;
  const int hi = lane >> 4, l15 = lane & 15;
  const int bh = blockIdx.y;
  const int b  = bh >> 4, h = bh & (H_ - 1);
  const int q0 = blockIdx.x * 64;
  const size_t headBase = (size_t)bh * S_ * DK_;

  // Q fragments (2 K-chunks of 32), loaded straight from global
  v16bf qf[2];
  {
    const __bf16* qrow = Q + headBase + (size_t)(q0 + wid * 16 + l15) * DK_;
#pragma unroll
    for (int c = 0; c < 2; ++c) {
      v8bf lo = *reinterpret_cast<const v8bf*>(qrow + c * 32 + hi * 8);
      v8bf hh = *reinterpret_cast<const v8bf*>(qrow + c * 32 + 16 + hi * 8);
#pragma unroll
      for (int j = 0; j < 8; ++j) { qf[c][j] = lo[j]; qf[c][8 + j] = hh[j]; }
    }
  }

  float mrow[8], lrow[8];
  v8f acc[4];
#pragma unroll
  for (int r = 0; r < 8; ++r) { mrow[r] = -1e30f; lrow[r] = 0.f; }
#pragma unroll
  for (int nd = 0; nd < 4; ++nd)
#pragma unroll
    for (int r = 0; r < 8; ++r) acc[nd][r] = 0.f;

  for (int k0 = 0; k0 < S_; k0 += 64) {
    __syncthreads();   // all waves done reading previous tiles
    // K tile: async global->LDS, row-major
#pragma unroll
    for (int it = 0; it < 4; ++it) {
      const int idx = it * 128 + tid;      // 0..511
      const int row = idx >> 3, c = idx & 7;
      async_copy16(&sK[row * VSTR + c * 8],
                   K + headBase + (size_t)(k0 + row) * DK_ + c * 8);
      // V tile: load and transpose into sVT (needs VGPR round-trip)
      v8bf vv = *reinterpret_cast<const v8bf*>(
          V + headBase + (size_t)(k0 + row) * DK_ + c * 8);
#pragma unroll
      for (int j = 0; j < 8; ++j) sVT[(c * 8 + j) * VSTR + row] = vv[j];
    }
    // prefetch next tiles toward WGP while this one is consumed
    if (k0 + 64 < S_) {
      __builtin_prefetch((const char*)(K + headBase + (size_t)(k0 + 64) * DK_) + tid * 32, 0, 0);
      __builtin_prefetch((const char*)(V + headBase + (size_t)(k0 + 64) * DK_) + tid * 32, 0, 0);
    }
    async_wait0();
    __syncthreads();

    // ---- scores S = (Q K^T) * 1/sqrt(DK) + mask ----
    v8f s[4];
#pragma unroll
    for (int ns = 0; ns < 4; ++ns) {
#pragma unroll
      for (int r = 0; r < 8; ++r) s[ns][r] = 0.f;
#pragma unroll
      for (int c = 0; c < 2; ++c) {
        v16bf kf;
        const __bf16* base = &sK[(ns * 16 + l15) * VSTR + c * 32 + hi * 8];
        v8bf b0 = *reinterpret_cast<const v8bf*>(base);
        v8bf b1 = *reinterpret_cast<const v8bf*>(base + 16);
#pragma unroll
        for (int j = 0; j < 8; ++j) { kf[j] = b0[j]; kf[8 + j] = b1[j]; }
        s[ns] = __builtin_amdgcn_wmma_f32_16x16x32_bf16(
            false, qf[c], false, kf, (short)0, s[ns], false, false);
      }
      const int key = k0 + ns * 16 + l15;
      const float madd = mask[b * S_ + key] ? 0.f : -1.0e9f;
#pragma unroll
      for (int r = 0; r < 8; ++r) s[ns][r] = s[ns][r] * 0.125f + madd;
    }

    // ---- online softmax (rows live in 16-lane halves) ----
    float mnew[8], psum[8], alpha[8];
#pragma unroll
    for (int r = 0; r < 8; ++r) {
      float mx = fmaxf(fmaxf(s[0][r], s[1][r]), fmaxf(s[2][r], s[3][r]));
#pragma unroll
      for (int off = 1; off < 16; off <<= 1) mx = fmaxf(mx, __shfl_xor(mx, off, 32));
      mnew[r]  = fmaxf(mrow[r], mx);
      alpha[r] = __expf(mrow[r] - mnew[r]);
      psum[r]  = 0.f;
    }
#pragma unroll
    for (int ns = 0; ns < 4; ++ns)
#pragma unroll
      for (int r = 0; r < 8; ++r) {
        const float p = __expf(s[ns][r] - mnew[r]);
        psum[r] += p;
        sP[wid][(hi * 8 + r) * 64 + ns * 16 + l15] = (__bf16)p;
      }
#pragma unroll
    for (int r = 0; r < 8; ++r) {
      float ps = psum[r];
#pragma unroll
      for (int off = 1; off < 16; off <<= 1) ps += __shfl_xor(ps, off, 32);
      lrow[r] = lrow[r] * alpha[r] + ps;
      mrow[r] = mnew[r];
    }
#pragma unroll
    for (int nd = 0; nd < 4; ++nd)
#pragma unroll
      for (int r = 0; r < 8; ++r) acc[nd][r] *= alpha[r];

    // ---- ctx += P * V  (P re-read from LDS in A layout, V from sVT) ----
#pragma unroll
    for (int c = 0; c < 2; ++c) {
      v16bf pf;
      const __bf16* pb = &sP[wid][l15 * 64 + c * 32 + hi * 8];
      v8bf lo = *reinterpret_cast<const v8bf*>(pb);
      v8bf hh = *reinterpret_cast<const v8bf*>(pb + 16);
#pragma unroll
      for (int j = 0; j < 8; ++j) { pf[j] = lo[j]; pf[8 + j] = hh[j]; }
#pragma unroll
      for (int nd = 0; nd < 4; ++nd) {
        v16bf vf;
        const __bf16* vb = &sVT[(nd * 16 + l15) * VSTR + c * 32 + hi * 8];
        v8bf v0 = *reinterpret_cast<const v8bf*>(vb);
        v8bf v1 = *reinterpret_cast<const v8bf*>(vb + 16);
#pragma unroll
        for (int j = 0; j < 8; ++j) { vf[j] = v0[j]; vf[8 + j] = v1[j]; }
        acc[nd] = __builtin_amdgcn_wmma_f32_16x16x32_bf16(
            false, pf, false, vf, (short)0, acc[nd], false, false);
      }
    }
  }

  // ---- normalize and store ctx as bf16 in [B,S,D] ----
#pragma unroll
  for (int nd = 0; nd < 4; ++nd)
#pragma unroll
    for (int r = 0; r < 8; ++r) {
      const int row = q0 + wid * 16 + hi * 8 + r;
      const float o = acc[nd][r] / lrow[r];
      ctx[((size_t)b * S_ + row) * D_ + h * DK_ + nd * 16 + l15] = (__bf16)o;
    }
}

// ---------------------------------------------------------------------------
// Launch
// ---------------------------------------------------------------------------
extern "C" void kernel_launch(void* const* d_in, const int* in_sizes, int n_in,
                              void* d_out, int out_size, void* d_ws, size_t ws_size,
                              hipStream_t stream)
{
  (void)in_sizes; (void)n_in; (void)out_size; (void)ws_size;

  const float* query = (const float*)d_in[0];
  const float* key   = (const float*)d_in[1];
  const float* value = (const float*)d_in[2];
  const unsigned char* mask = (const unsigned char*)d_in[3];
  const float* wq = (const float*)d_in[4];
  const float* bq = (const float*)d_in[5];
  const float* wk = (const float*)d_in[6];
  const float* bk = (const float*)d_in[7];
  const float* wv = (const float*)d_in[8];
  const float* bv = (const float*)d_in[9];
  const float* wo = (const float*)d_in[10];
  const float* bo = (const float*)d_in[11];

  char* ws = (char*)d_ws;
  const size_t MB = 1024u * 1024u;
  __bf16* xq   = (__bf16*)(ws + 0 * MB);    // 4096x1024 bf16 = 8MB
  __bf16* xk   = (__bf16*)(ws + 8 * MB);
  __bf16* xv   = (__bf16*)(ws + 16 * MB);
  __bf16* wqb  = (__bf16*)(ws + 24 * MB);   // 1024x1024 bf16 = 2MB
  __bf16* wkb  = (__bf16*)(ws + 26 * MB);
  __bf16* wvb  = (__bf16*)(ws + 28 * MB);
  __bf16* wob  = (__bf16*)(ws + 30 * MB);
  __bf16* Qh   = (__bf16*)(ws + 32 * MB);   // [B,H,S,DK] bf16 = 8MB
  __bf16* Kh   = (__bf16*)(ws + 40 * MB);
  __bf16* Vh   = (__bf16*)(ws + 48 * MB);
  __bf16* ctxb = (__bf16*)(ws + 56 * MB);   // [B,S,D] bf16 = 8MB

  const int nx = MTOT * D_;   // 4194304
  const int nw = D_ * D_;     // 1048576
  cvt_f32_bf16<<<nx / 256, 256, 0, stream>>>(query, xq, nx);
  cvt_f32_bf16<<<nx / 256, 256, 0, stream>>>(key,   xk, nx);
  cvt_f32_bf16<<<nx / 256, 256, 0, stream>>>(value, xv, nx);
  cvt_f32_bf16<<<nw / 256, 256, 0, stream>>>(wq, wqb, nw);
  cvt_f32_bf16<<<nw / 256, 256, 0, stream>>>(wk, wkb, nw);
  cvt_f32_bf16<<<nw / 256, 256, 0, stream>>>(wv, wvb, nw);
  cvt_f32_bf16<<<nw / 256, 256, 0, stream>>>(wo, wob, nw);

  dim3 gGrid(D_ / 128, MTOT / 128);  // (8, 32)
  gemm_bf16_wmma<0><<<gGrid, 256, 0, stream>>>(xq, wqb, bq, Qh, MTOT, D_, D_);
  gemm_bf16_wmma<0><<<gGrid, 256, 0, stream>>>(xk, wkb, bk, Kh, MTOT, D_, D_);
  gemm_bf16_wmma<0><<<gGrid, 256, 0, stream>>>(xv, wvb, bv, Vh, MTOT, D_, D_);

  dim3 fGrid(S_ / 64, B_ * H_);      // (32, 32)
  flash_attn_wmma<<<fGrid, 128, 0, stream>>>(Qh, Kh, Vh, mask, ctxb);

  gemm_bf16_wmma<1><<<gGrid, 256, 0, stream>>>(ctxb, wob, bo, d_out, MTOT, D_, D_);
}
